// VQCodebook_4681514352959
// MI455X (gfx1250) — matrix operand order repaired and checked by reading, not compiled, not run on hardware
//
#include <hip/hip_runtime.h>
#include <hip/hip_bf16.h>

// ---------------------------------------------------------------------------
// VQ codebook lookup on MI455X (gfx1250, wave32).
//   scores = |e_k|^2 - 2 * z . e_k   (|z|^2 is constant per row -> dropped)
//   GEMM 32768 x 8192 x 512 in bf16 WMMA (f32 accumulate) with fused argmin.
//   B tiles double-buffered in LDS via GLOBAL_LOAD_ASYNC_TO_LDS_B128.
// ---------------------------------------------------------------------------

#define HDIM   512
#define KCODES 8192
#define NTOK   32768           // B*T = 8*4096
#define MWG    256             // tokens per workgroup (A tile)
#define NGRP   32              // codes per LDS B tile (2 N-tiles)
#define NGROUPS (KCODES / NGRP)

typedef __attribute__((ext_vector_type(16))) __bf16 bf16x16;
typedef __attribute__((ext_vector_type(8)))  float  f32x8;

union Frag { uint4 q[2]; bf16x16 v; };

__device__ __forceinline__ unsigned int f2bf(float f) {
    unsigned int u = __float_as_uint(f);
    u += 0x7fffu + ((u >> 16) & 1u);      // round-to-nearest-even
    return u >> 16;
}

__device__ __forceinline__ void wait_async0() {
#if __has_builtin(__builtin_amdgcn_s_wait_asynccnt)
    __builtin_amdgcn_s_wait_asynccnt(0);
#else
    asm volatile("s_wait_asynccnt 0x0" ::: "memory");
#endif
}

// ---------------------------------------------------------------------------
// Kernel 1: z (f32) -> z_bf16.  grid = 16384 x 256, one float4 per thread.
// ---------------------------------------------------------------------------
__global__ __launch_bounds__(256) void zconv_kernel(const float* __restrict__ z,
                                                    unsigned short* __restrict__ zb) {
    size_t i = (size_t)blockIdx.x * 256 + threadIdx.x;     // float4 index
    float4 v = ((const float4*)z)[i];
    uint2 o;
    o.x = f2bf(v.x) | (f2bf(v.y) << 16);
    o.y = f2bf(v.z) | (f2bf(v.w) << 16);
    ((uint2*)zb)[i] = o;
}

// ---------------------------------------------------------------------------
// Kernel 2: emb (f32) -> emb_bf16 and e_norms (exact f32 row norms).
// One wave per codebook row.  grid = 1024 x 256 (8 waves/block).
// ---------------------------------------------------------------------------
__global__ __launch_bounds__(256) void embprep_kernel(const float* __restrict__ emb,
                                                      unsigned short* __restrict__ eb,
                                                      float* __restrict__ enorm) {
    int wave = threadIdx.x >> 5, lane = threadIdx.x & 31;
    int code = blockIdx.x * 8 + wave;
    const float4* src = (const float4*)(emb + (size_t)code * HDIM);
    uint2* dst = (uint2*)(eb + (size_t)code * HDIM);
    float s = 0.f;
#pragma unroll
    for (int it = 0; it < 4; ++it) {
        float4 v = src[lane + it * 32];
        s += v.x * v.x + v.y * v.y + v.z * v.z + v.w * v.w;
        uint2 o;
        o.x = f2bf(v.x) | (f2bf(v.y) << 16);
        o.y = f2bf(v.z) | (f2bf(v.w) << 16);
        dst[lane + it * 32] = o;
    }
#pragma unroll
    for (int off = 16; off; off >>= 1) s += __shfl_down(s, off, 32);
    if (lane == 0) enorm[code] = s;
}

// ---------------------------------------------------------------------------
// Kernel 3: fused GEMM + argmin.
// A (256 tok x 512) staged once in LDS, pre-permuted into WMMA A-fragment
// order; emb streamed as 32-code tiles, double-buffered in LDS via
// GLOBAL_LOAD_ASYNC_TO_LDS_B128 (ASYNCcnt) so the next tile's DMA overlaps
// the current tile's 16 k-step WMMA phase.  Each wave: 2 M-tiles x 2 N-tiles,
// v_wmma_f32_16x16x32_bf16, running per-lane argmin, final cross-lane reduce
// via __shfl_xor over 16-lane column groups.
// Dynamic LDS: 16384 + 2*2048 uint4 = 327680 B (= 320 KB WGP limit).
// grid = 128 x 256.
// ---------------------------------------------------------------------------
__global__ __launch_bounds__(256) void vq_argmin_kernel(
        const unsigned short* __restrict__ zb,     // [NTOK][HDIM] bf16
        const unsigned short* __restrict__ eb,     // [KCODES][HDIM] bf16
        const float* __restrict__ enorm,           // [KCODES]
        int* __restrict__ bestidx) {               // [NTOK]
    extern __shared__ uint4 smem[];
    uint4* sA = smem;                 // 16 mtiles x 16 kc x 32 lanes x 2 uint4
    uint4* sB = smem + 16384;         // 2 bufs x (2 ntiles x 16 kc x 32 x 2)

    const int tid  = threadIdx.x;
    const int wave = tid >> 5;
    const int lane = tid & 31;
    const int hlf  = lane >> 4;       // 0: rows r, 1: rows r+8 in C tile
    const int col  = lane & 15;       // C column within N-tile
    const int wgbase = blockIdx.x * MWG;
    const int mt0 = wave * 2;         // this wave's M-tiles: mt0, mt0+1

    // ---- stage A tile (256 tokens x 512 h) into fragment-permuted LDS ----
    {
        const uint4* gz = (const uint4*)zb + (size_t)wgbase * (HDIM / 8);
        for (int j = tid; j < MWG * HDIM / 8; j += 256) {
            int t  = j >> 6;                 // token in WG (64 uint4 per row)
            int k8 = (j & 63) << 3;          // h base of this uint4
            int mtile = t >> 4, m = t & 15;
            int kc = k8 >> 5, kin = k8 & 31; // 32-wide k chunk
            int hh = (kin >> 3) & 1;         // A half: K blocks {0,16}->0, {8,24}->1
            int e0 = (kin >= 16) ? 1 : 0;    // low/high uint4 of the fragment
            int ln = m + (hh << 4);
            sA[((((mtile << 4) + kc) << 5) + ln) * 2 + e0] = gz[j];
        }
    }

    // ---- async staging of one 32-code B tile into buffer b (permuted) ----
    auto issueB = [&](int g, int b) {
        unsigned gbase = (unsigned)g * (NGRP * HDIM * 2);   // byte offset in eb
#pragma unroll
        for (int j = tid; j < NGRP * HDIM / 8; j += 256) {
            int n  = j >> 6;
            int k8 = (j & 63) << 3;
            int nt = n >> 4, nn = n & 15;
            int kc = k8 >> 5, kin = k8 & 31;
            int hh = (kin >> 3) & 1;             // B half: K blocks {0,16}->0, {8,24}->1
            int e0 = (kin >= 16) ? 1 : 0;
            int ln = nn + (hh << 4);
            unsigned elem = (unsigned)(((((nt << 4) + kc) << 5) + ln) * 2 + e0);
            // dynamic LDS base is 0 in this kernel (no static __shared__)
            unsigned ldsa = (16384u + (unsigned)(b * 2048) + elem) * 16u;
            unsigned goff = gbase + (unsigned)j * 16u;
            asm volatile("global_load_async_to_lds_b128 %0, %1, %2"
                         :: "v"(ldsa), "v"(goff), "s"(eb) : "memory");
        }
    };

    float bestv[2][8];
    int   besti[2][8];
#pragma unroll
    for (int mt = 0; mt < 2; ++mt)
#pragma unroll
        for (int r = 0; r < 8; ++r) { bestv[mt][r] = 3.4e38f; besti[mt][r] = 0; }

    issueB(0, 0);

    for (int g = 0; g < NGROUPS; ++g) {
        wait_async0();       // this wave's async loads for buf[g&1] complete
        __syncthreads();     // all waves' loads + previous compute complete
        if (g + 1 < NGROUPS) issueB(g + 1, (g + 1) & 1);   // overlap with WMMA

        const uint4* bb = sB + (g & 1) * 2048;
        float en0 = enorm[g * NGRP + col];
        float en1 = enorm[g * NGRP + 16 + col];

        f32x8 acc00 = {}; f32x8 acc01 = {}; f32x8 acc10 = {}; f32x8 acc11 = {};
#pragma unroll 4
        for (int kc = 0; kc < 16; ++kc) {
            Frag a0, a1, b0, b1;
            const uint4* pa0 = &sA[(((mt0 << 4) + kc) << 5) * 2 + lane * 2];
            const uint4* pa1 = pa0 + (16 << 5) * 2;        // next M-tile
            a0.q[0] = pa0[0]; a0.q[1] = pa0[1];
            a1.q[0] = pa1[0]; a1.q[1] = pa1[1];
            const uint4* pb0 = bb + (kc << 5) * 2 + lane * 2;
            const uint4* pb1 = pb0 + (16 << 5) * 2;        // next N-tile
            b0.q[0] = pb0[0]; b0.q[1] = pb0[1];
            b1.q[0] = pb1[0]; b1.q[1] = pb1[1];

            acc00 = __builtin_amdgcn_wmma_f32_16x16x32_bf16(false, a0.v, false, b0.v,
                                                            (short)0, acc00, false, false);
            acc01 = __builtin_amdgcn_wmma_f32_16x16x32_bf16(false, a0.v, false, b1.v,
                                                            (short)0, acc01, false, false);
            acc10 = __builtin_amdgcn_wmma_f32_16x16x32_bf16(false, a1.v, false, b0.v,
                                                            (short)0, acc10, false, false);
            acc11 = __builtin_amdgcn_wmma_f32_16x16x32_bf16(false, a1.v, false, b1.v,
                                                            (short)0, acc11, false, false);
        }

        int i0 = g * NGRP + col;
        int i1 = i0 + 16;
#pragma unroll
        for (int r = 0; r < 8; ++r) {
            float s;
            bool c;
            s = en0 - 2.f * acc00[r];
            c = s < bestv[0][r]; bestv[0][r] = c ? s : bestv[0][r]; besti[0][r] = c ? i0 : besti[0][r];
            s = en1 - 2.f * acc01[r];
            c = s < bestv[0][r]; bestv[0][r] = c ? s : bestv[0][r]; besti[0][r] = c ? i1 : besti[0][r];
            s = en0 - 2.f * acc10[r];
            c = s < bestv[1][r]; bestv[1][r] = c ? s : bestv[1][r]; besti[1][r] = c ? i0 : besti[1][r];
            s = en1 - 2.f * acc11[r];
            c = s < bestv[1][r]; bestv[1][r] = c ? s : bestv[1][r]; besti[1][r] = c ? i1 : besti[1][r];
        }
    }

    // ---- reduce argmin across the 16 column lanes of each half-wave ----
#pragma unroll
    for (int mt = 0; mt < 2; ++mt) {
#pragma unroll
        for (int r = 0; r < 8; ++r) {
            float v = bestv[mt][r];
            int   i = besti[mt][r];
#pragma unroll
            for (int off = 1; off < 16; off <<= 1) {
                float ov = __shfl_xor(v, off, 32);
                int   oi = __shfl_xor(i, off, 32);
                if (ov < v || (ov == v && oi < i)) { v = ov; i = oi; }
            }
            if (col == 0) {
                int token = wgbase + (mt0 + mt) * 16 + r + hlf * 8;
                bestidx[token] = i;
            }
        }
    }
}

// ---------------------------------------------------------------------------
// Kernel 4: gather z_q = emb[idx], write z_st = z + (z_q - z), idx as float,
// per-block partial sums of (z_q - z)^2.  One wave per token, grid = 4096x256.
// ---------------------------------------------------------------------------
__global__ __launch_bounds__(256) void gather_kernel(
        const float* __restrict__ z, const float* __restrict__ emb,
        const int* __restrict__ bestidx,
        float* __restrict__ out_zst, float* __restrict__ out_idx,
        float* __restrict__ partials) {
    __shared__ float wsum[8];
    int wave = threadIdx.x >> 5, lane = threadIdx.x & 31;
    int token = blockIdx.x * 8 + wave;
    int idx = bestidx[token];
    if (lane == 0) out_idx[token] = (float)idx;

    const float4* zp = (const float4*)(z + (size_t)token * HDIM);
    const float4* ep = (const float4*)(emb + (size_t)idx * HDIM);
    float4* op = (float4*)(out_zst + (size_t)token * HDIM);
    float s = 0.f;
#pragma unroll
    for (int it = 0; it < 4; ++it) {
        float4 zv = zp[lane + it * 32];
        float4 ev = ep[lane + it * 32];
        float4 d, st;
        d.x = ev.x - zv.x; d.y = ev.y - zv.y; d.z = ev.z - zv.z; d.w = ev.w - zv.w;
        s += d.x * d.x + d.y * d.y + d.z * d.z + d.w * d.w;
        st.x = zv.x + d.x; st.y = zv.y + d.y; st.z = zv.z + d.z; st.w = zv.w + d.w;
        op[lane + it * 32] = st;
    }
#pragma unroll
    for (int off = 16; off; off >>= 1) s += __shfl_down(s, off, 32);
    if (lane == 0) wsum[wave] = s;
    __syncthreads();
    if (threadIdx.x == 0) {
        float t = 0.f;
#pragma unroll
        for (int w = 0; w < 8; ++w) t += wsum[w];
        partials[blockIdx.x] = t;
    }
}

// ---------------------------------------------------------------------------
// Kernel 5: deterministic reduce of 4096 partials -> both loss scalars.
// ---------------------------------------------------------------------------
__global__ __launch_bounds__(256) void loss_kernel(const float* __restrict__ partials,
                                                   float* __restrict__ out_loss) {
    __shared__ float red[256];
    int tid = threadIdx.x;
    float s = 0.f;
    for (int i = tid; i < 4096; i += 256) s += partials[i];
    red[tid] = s;
    __syncthreads();
    for (int off = 128; off; off >>= 1) {
        if (tid < off) red[tid] += red[tid + off];
        __syncthreads();
    }
    if (tid == 0) {
        float m = red[0] / (float)((double)NTOK * (double)HDIM);
        out_loss[0] = m;   // codebook_loss == commit_loss numerically
        out_loss[1] = m;
    }
}

// ---------------------------------------------------------------------------
extern "C" void kernel_launch(void* const* d_in, const int* in_sizes, int n_in,
                              void* d_out, int out_size, void* d_ws, size_t ws_size,
                              hipStream_t stream) {
    const float* z   = (const float*)d_in[0];   // (8,4096,512)
    const float* emb = (const float*)d_in[1];   // (8192,512)

    char* ws = (char*)d_ws;
    unsigned short* z_bf16   = (unsigned short*)(ws + 0);            // 32 MiB
    unsigned short* emb_bf16 = (unsigned short*)(ws + 33554432);     //  8 MiB
    float*          e_norms  = (float*)(ws + 41943040);              // 32 KiB
    int*            best_idx = (int*)(ws + 41975808);                // 128 KiB
    float*          partials = (float*)(ws + 42106880);              // 16 KiB

    float* out   = (float*)d_out;
    float* o_zst = out;                       // 16777216 floats
    float* o_idx = out + (size_t)NTOK * HDIM; // 32768 floats
    float* o_los = o_idx + NTOK;              // 2 floats

    zconv_kernel<<<dim3((NTOK * HDIM / 4) / 256), dim3(256), 0, stream>>>(z, z_bf16);
    embprep_kernel<<<dim3(KCODES / 8), dim3(256), 0, stream>>>(emb, emb_bf16, e_norms);
    vq_argmin_kernel<<<dim3(NTOK / MWG), dim3(256), 327680, stream>>>(
        z_bf16, emb_bf16, e_norms, best_idx);
    gather_kernel<<<dim3(NTOK / 8), dim3(256), 0, stream>>>(
        z, emb, best_idx, o_zst, o_idx, partials);
    loss_kernel<<<dim3(1), dim3(256), 0, stream>>>(partials, o_los);
}
